// Dynamic_deformable_DySamplev6_40338332844083
// MI455X (gfx1250) — matrix operand order repaired and checked by reading, not compiled, not run on hardware
//
#include <hip/hip_runtime.h>
#include <math.h>

// ---------------- problem constants ----------------
#define BB   4
#define HH   352
#define WW   1216
#define CG   27      // guidance channels (CH_G * PROP_TIME)
#define CO   81      // conv output channels (3 * CG)
#define KTAP 9       // 3x3 taps
#define PT   3       // PROP_TIME

#define KP   256     // K padded (27*9 = 243 -> 256)
#define MP   96      // M padded (81 -> 96 = 6 WMMA tiles)
#define TILE_N 64    // pixels per block (4 x 16-wide WMMA N-tiles)
#define RAWW 66      // raw guidance cols staged (64 + 2 halo)
#define RAWP 68      // padded stride for LDS bank spread

#define HW   ((size_t)HH * WW)
#define BHW  ((size_t)BB * HW)

// d_out layout (floats, reference return order, concatenated flat):
//   [0, BHW)                         final feat
//   [BHW, 4*BHW)                     list_feat (3 steps)
//   [4*BHW,  4*BHW + 54*BHW)         offsets (B,3,18,H,W), f16-rounded
//   [58*BHW, 58*BHW + 27*BHW)        affs    (B,3, 9,H,W), softmaxed, f16-rounded
#define OUT_FEAT   ((size_t)0)
#define OUT_LIST   (BHW)
#define OUT_OFFS   ((size_t)4 * BHW)
#define OUT_AFFS   (OUT_OFFS + (size_t)54 * BHW)

// workspace layout (bytes)
#define WS_WPK   ((size_t)0)                         // MP*KP f16 packed weights
#define WS_CONF  ((size_t)(MP * KP * 2 + 256))       // BHW f32 conf
#define WS_FEATA (WS_CONF  + BHW * 4)                // BHW f32 feat ping
#define WS_FEATB (WS_FEATA + BHW * 4)                // BHW f32 feat pong

typedef __attribute__((ext_vector_type(16))) _Float16 v16h;
typedef __attribute__((ext_vector_type(8)))  _Float16 v8h;
typedef __attribute__((ext_vector_type(8)))  float    v8f;

// ---------------------------------------------------------------------------
// Kernel 0: pack conv weights (OIHW f32) -> WMMA-A f16 [MP][KP],
// k = (ky*3+kx)*CG + c, zero padded.
// ---------------------------------------------------------------------------
__global__ __launch_bounds__(256) void pack_weights(const float* __restrict__ w_off,
                                                    _Float16* __restrict__ wpk) {
    int i = blockIdx.x * 256 + threadIdx.x;      // over MP*KP = 24576
    if (i >= MP * KP) return;
    int o = i / KP, k = i % KP;
    float v = 0.f;
    if (o < CO && k < CG * KTAP) {
        int tap = k / CG, c = k - tap * CG;
        int ky = tap / 3, kx = tap - ky * 3;
        v = w_off[(((size_t)o * CG + c) * 3 + ky) * 3 + kx];
    }
    wpk[i] = (_Float16)v;
}

// ---------------------------------------------------------------------------
// Kernel 1: implicit-GEMM 3x3 conv via v_wmma_f32_16x16x32_f16.
// Block = (b, h, 64-pixel strip); 6 waves x 16 out-channels = 96 (81 live).
// A fragments held in registers across 4 N-tiles; guidance + weights staged
// to LDS with GLOBAL_LOAD_ASYNC_TO_LDS (ASYNCcnt path).
// ---------------------------------------------------------------------------
__global__ __launch_bounds__(192) void conv_offsets_wmma(const float* __restrict__ guid,
                                                         const float* __restrict__ b_off,
                                                         const _Float16* __restrict__ wpk,
                                                         float* __restrict__ out) {
    __shared__ _Float16 Asm[MP * KP];         // 48 KB weights (A)
    __shared__ _Float16 Bsm[TILE_N * KP];     // 32 KB im2col (B)
    __shared__ float    Rsm[CG * 3 * RAWP];   // ~21.5 KB raw guidance rows

    const int tid = threadIdx.x;
    int blk = blockIdx.x;
    const int wt = blk % (WW / TILE_N); blk /= (WW / TILE_N);
    const int h  = blk % HH;            const int b = blk / HH;
    const int w0 = wt * TILE_N;

    // --- async stage: packed weights (contiguous 48 KB, B128 granularity) ---
    for (int i = tid; i < (MP * KP * 2) / 16; i += 192) {
        uint32_t ldsoff = (uint32_t)(uintptr_t)(&Asm[0]) + (uint32_t)i * 16u;
        uint32_t goff   = (uint32_t)i * 16u;
        asm volatile("global_load_async_to_lds_b128 %0, %1, %2"
                     :: "v"(ldsoff), "v"(goff), "s"(wpk) : "memory");
    }
    // --- async stage: raw guidance rows (halo'd strip), zero-fill OOB ---
    for (int t = tid; t < CG * 3 * RAWW; t += 192) {
        const int j = t % RAWW;
        const int rc = t / RAWW;
        const int r = rc % 3, c = rc / 3;
        const int gy = h - 1 + r, gx = w0 - 1 + j;
        const int lidx = (c * 3 + r) * RAWP + j;
        if (gy >= 0 && gy < HH && gx >= 0 && gx < WW) {
            uint32_t ldsoff = (uint32_t)(uintptr_t)(&Rsm[lidx]);
            uint32_t goff   = (uint32_t)(((((size_t)b * CG + c) * HH + gy) * WW + gx) * 4);
            asm volatile("global_load_async_to_lds_b32 %0, %1, %2"
                         :: "v"(ldsoff), "v"(goff), "s"(guid) : "memory");
        } else {
            Rsm[lidx] = 0.f;
        }
    }
    asm volatile("s_wait_asynccnt 0" ::: "memory");
    __syncthreads();

    // --- build im2col f16 B [TILE_N][KP] from raw LDS ---
    for (int i = tid; i < TILE_N * KP; i += 192) {
        const int n = i >> 8, k = i & (KP - 1);
        float v = 0.f;
        if (k < CG * KTAP) {
            const int tap = k / CG, c = k - tap * CG;
            const int ky = tap / 3, kx = tap - ky * 3;
            v = Rsm[(c * 3 + ky) * RAWP + n + kx];
        }
        Bsm[n * KP + k] = (_Float16)v;
    }
    __syncthreads();

    const int wave = tid >> 5, lane = tid & 31;
    const int lo = lane & 15, hi = lane >> 4;

    // A fragments: load once, keep in registers (8 x v16h = 64 VGPRs)
    const _Float16* Arow = &Asm[(wave * 16 + lo) * KP];
    v16h afr[8];
    #pragma unroll
    for (int kk = 0; kk < 8; ++kk) {
        union { v16h v; v8h h[2]; } a;
        a.h[0] = *(const v8h*)(Arow + kk * 32 + hi * 8);
        a.h[1] = *(const v8h*)(Arow + kk * 32 + 16 + hi * 8);
        afr[kk] = a.v;
    }

    // 4 N-tiles of 16 pixels each; 8 WMMAs per tile
    #pragma unroll
    for (int nt = 0; nt < 4; ++nt) {
        const _Float16* Brow = &Bsm[(nt * 16 + lo) * KP];
        v8f acc = {};
        #pragma unroll
        for (int kk = 0; kk < 8; ++kk) {
            union { v16h v; v8h h[2]; } bm;
            bm.h[0] = *(const v8h*)(Brow + kk * 32 + hi * 8);
            bm.h[1] = *(const v8h*)(Brow + kk * 32 + 16 + hi * 8);
            acc = __builtin_amdgcn_wmma_f32_16x16x32_f16(false, afr[kk], false, bm.v,
                                                         (short)0, acc, false, false);
        }
        // C layout: pixel = lane%16; channel = wave*16 + r + (lane>=16 ? 8 : 0)
        const size_t pix = (size_t)h * WW + w0 + nt * 16 + lo;
        #pragma unroll
        for (int r = 0; r < 8; ++r) {
            const int o = wave * 16 + r + hi * 8;
            if (o < CO) {
                const float val = acc[r] + b_off[o];
                const int prop  = o / CG;
                const int wi    = o - prop * CG;
                if (wi < 2 * KTAP) {    // offset channel: round to f16 now
                    out[OUT_OFFS + ((size_t)((b * PT + prop) * 18 + wi)) * HW + pix] =
                        (float)(_Float16)val;
                } else {                // affinity logit: softmaxed in pass 2
                    out[OUT_AFFS + ((size_t)((b * PT + prop) * 9 + (wi - 18))) * HW + pix] = val;
                }
            }
        }
    }
}

// ---------------------------------------------------------------------------
// Kernel 2a: conf = sign(feat_fix) * sigmoid(confidence)
// ---------------------------------------------------------------------------
__global__ __launch_bounds__(256) void conf_kernel(const float* __restrict__ confidence,
                                                   const float* __restrict__ feat_fix,
                                                   float* __restrict__ conf) {
    size_t i = (size_t)blockIdx.x * 256 + threadIdx.x;
    if (i >= BHW) return;
    const float x = feat_fix[i];
    const float s = (x > 0.f) ? 1.f : ((x < 0.f) ? -1.f : 0.f);
    conf[i] = s * (1.f / (1.f + __expf(-confidence[i])));
}

// ---------------------------------------------------------------------------
// Kernel 2b: softmax over 9 affinity logits (in-place in d_out), f16 round
// ---------------------------------------------------------------------------
__global__ __launch_bounds__(256) void softmax_affs(float* __restrict__ out) {
    size_t i = (size_t)blockIdx.x * 256 + threadIdx.x;   // over B*PT*HW
    if (i >= (size_t)BB * PT * HW) return;
    const size_t pix = i % HW, g = i / HW;               // g = b*PT + prop
    float* base = out + OUT_AFFS + (size_t)g * 9 * HW + pix;
    float l[9], m = -3.4e38f;
    #pragma unroll
    for (int j = 0; j < 9; ++j) { l[j] = base[j * HW]; m = fmaxf(m, l[j]); }
    float s = 0.f;
    #pragma unroll
    for (int j = 0; j < 9; ++j) { l[j] = __expf(l[j] - m); s += l[j]; }
    const float inv = 1.f / s;
    #pragma unroll
    for (int j = 0; j < 9; ++j) base[j * HW] = (float)(_Float16)(l[j] * inv);
}

// ---------------------------------------------------------------------------
// Kernel 3: one deformable-propagation step (9-tap bilinear gather + blend)
// ---------------------------------------------------------------------------
__global__ __launch_bounds__(256) void prop_step(const float* __restrict__ fin,
                                                 float* __restrict__ fout,
                                                 const float* __restrict__ conf,
                                                 const float* __restrict__ feat_fix,
                                                 const float* __restrict__ w_f,
                                                 const float* __restrict__ b_f,
                                                 const float* __restrict__ outro,
                                                 float* __restrict__ out,
                                                 int kstep) {
    size_t i = (size_t)blockIdx.x * 256 + threadIdx.x;
    if (i >= BHW) return;
    const size_t pix = i % HW;
    const int b = (int)(i / HW);
    const int h = (int)(pix / WW), w = (int)(pix % WW);
    const float* fplane = fin + (size_t)b * HW;
    const size_t obase = OUT_OFFS + ((size_t)(b * PT + kstep) * 18) * HW + pix;
    const size_t abase = OUT_AFFS + ((size_t)(b * PT + kstep) * 9) * HW + pix;

    float sum = 0.f;
    #pragma unroll
    for (int tap = 0; tap < 9; ++tap) {
        const float dy = (float)(tap / 3 - 1), dx = (float)(tap % 3 - 1);
        const float ys = (float)h + dy + outro[obase + (size_t)(tap * 2)     * HW];
        const float xs = (float)w + dx + outro[obase + (size_t)(tap * 2 + 1) * HW];
        const float y0 = floorf(ys), x0 = floorf(xs);
        const float wy = ys - y0,    wx = xs - x0;
        const int y0i = (int)y0, x0i = (int)x0;
        float s00 = 0.f, s01 = 0.f, s10 = 0.f, s11 = 0.f;
        if (y0i >= 0 && y0i < HH) {
            if (x0i >= 0 && x0i < WW)         s00 = fplane[(size_t)y0i * WW + x0i];
            if (x0i + 1 >= 0 && x0i + 1 < WW) s01 = fplane[(size_t)y0i * WW + x0i + 1];
        }
        if (y0i + 1 >= 0 && y0i + 1 < HH) {
            if (x0i >= 0 && x0i < WW)         s10 = fplane[(size_t)(y0i + 1) * WW + x0i];
            if (x0i + 1 >= 0 && x0i + 1 < WW) s11 = fplane[(size_t)(y0i + 1) * WW + x0i + 1];
        }
        const float v = s00 * (1.f - wy) * (1.f - wx) + s01 * (1.f - wy) * wx +
                        s10 * wy * (1.f - wx)         + s11 * wy * wx;
        sum += v * outro[abase + (size_t)tap * HW] * w_f[tap];
    }
    const float propv = sum + b_f[0];
    const float c = conf[i];
    const float feat = (1.f - c) * propv + c * feat_fix[i];
    fout[i] = feat;
    out[OUT_LIST + (size_t)kstep * BHW + i] = feat;
    if (kstep == PT - 1) out[OUT_FEAT + i] = feat;
}

// ---------------------------------------------------------------------------
extern "C" void kernel_launch(void* const* d_in, const int* in_sizes, int n_in,
                              void* d_out, int out_size, void* d_ws, size_t ws_size,
                              hipStream_t stream) {
    const float* feat_init  = (const float*)d_in[0];
    const float* guidance   = (const float*)d_in[1];
    const float* confidence = (const float*)d_in[2];
    const float* feat_fix   = (const float*)d_in[3];
    const float* w_off      = (const float*)d_in[4];
    const float* b_off      = (const float*)d_in[5];
    const float* w_f        = (const float*)d_in[6];
    const float* b_f        = (const float*)d_in[7];
    float* out = (float*)d_out;

    char* ws = (char*)d_ws;
    _Float16* wpk  = (_Float16*)(ws + WS_WPK);
    float*    conf = (float*)(ws + WS_CONF);
    float*    fA   = (float*)(ws + WS_FEATA);
    float*    fB   = (float*)(ws + WS_FEATB);

    // 0: pack weights to f16 A-layout
    pack_weights<<<(MP * KP + 255) / 256, 256, 0, stream>>>(w_off, wpk);
    // 1: WMMA conv -> offsets (f16-rounded) + affinity logits into d_out
    const int nblk = BB * HH * (WW / TILE_N);
    conv_offsets_wmma<<<nblk, 192, 0, stream>>>(guidance, b_off, wpk, out);
    // 2: conf and softmax
    conf_kernel<<<(int)((BHW + 255) / 256), 256, 0, stream>>>(confidence, feat_fix, conf);
    softmax_affs<<<(int)(((size_t)BB * PT * HW + 255) / 256), 256, 0, stream>>>(out);
    // 3: three propagation steps (feat_init -> fA -> fB -> fA)
    const int pblk = (int)((BHW + 255) / 256);
    prop_step<<<pblk, 256, 0, stream>>>(feat_init, fA, conf, feat_fix, w_f, b_f, out, out, 0);
    prop_step<<<pblk, 256, 0, stream>>>(fA,        fB, conf, feat_fix, w_f, b_f, out, out, 1);
    prop_step<<<pblk, 256, 0, stream>>>(fB,        fA, conf, feat_fix, w_f, b_f, out, out, 2);
}
